// MambaDecoder_82420422410416
// MI455X (gfx1250) — compile-verified, matrix operand order checked by reading, not run on hardware
//
#include <hip/hip_runtime.h>
#include <hip/hip_bf16.h>

// ---------------------------------------------------------------------------
// MambaDecoder for MI455X (gfx1250): bf16 WMMA GEMMs (2x2 register blocking)
// + fused epilogues, sequential Mamba scan, batched WMMA attention.
// ---------------------------------------------------------------------------

typedef __attribute__((ext_vector_type(16))) __bf16        v16bf;
typedef __attribute__((ext_vector_type(8)))  float         v8f;
typedef __attribute__((ext_vector_type(4)))  float         f32x4;
typedef __attribute__((ext_vector_type(4)))  unsigned int  u32x4;

union Frag { v16bf v; u32x4 q[2]; };

#define DMODEL 1024
#define DI     2048       // D * EXP
#define NDST   16         // DSTATE
#define NHEADS 8
#define HDIM   128
#define BATCH  4
#define SEQ    1024
#define NTOK   (BATCH * SEQ)   // 4096
#define LDSS   40              // LDS row stride in bf16 elements (80B, 16B aligned)

// ---------------------------------------------------------------------------
// Generic WMMA GEMM:  Out = act(scale * A @ W^T + bias) + resScale * Res
//   A: [M,K] rows stride ldA ; W: [N,K] rows stride ldW ; Out: [M,N] stride ldO
//   Batched via grid.z: z -> (zb = z/zDiv, zh = z%zDiv), per-operand strides.
//   Block: 256 threads (8 waves) -> 64x128 tile; each wave computes a 32x32
//   region as 2x2 WMMA tiles (4 x v_wmma per K-step of 32).
//   act: 0 = none, 1 = SiLU, 2 = exact GELU.  Requires M%64==0, N%128==0, K%32==0.
// ---------------------------------------------------------------------------
__global__ __launch_bounds__(256)
void wmma_gemm_kernel(const float* __restrict__ A, int ldA, long long aSb, long long aSh,
                      const float* __restrict__ W, int ldW, long long wSb, long long wSh,
                      const float* __restrict__ bias,
                      float* __restrict__ Out, int ldO, long long oSb, long long oSh,
                      const float* __restrict__ Res, float resScale,
                      int M, int N, int K, float scale, int act, int zDiv)
{
    __shared__ __bf16 aT[64  * LDSS];   //  5.0 KB
    __shared__ __bf16 wT[128 * LDSS];   // 10.0 KB

    const int z  = blockIdx.z;
    const int zb = z / zDiv, zh = z % zDiv;
    A   += (long long)zb * aSb + (long long)zh * aSh;
    W   += (long long)zb * wSb + (long long)zh * wSh;
    Out += (long long)zb * oSb + (long long)zh * oSh;

    const int tid  = threadIdx.x;
    const int lane = tid & 31;
    const int wv   = tid >> 5;
    const int wm   = wv >> 2;          // 0..1  (32-row half of block tile)
    const int wn   = wv & 3;           // 0..3  (32-col quarter)
    const int l    = lane & 15;
    const int hi   = lane >> 4;

    const int m0 = blockIdx.y << 6;    // * 64
    const int n0 = blockIdx.x << 7;    // * 128

    // cooperative-load mapping
    const int aRow = tid >> 2;             // 0..63
    const int aCol = (tid & 3) << 3;       // 0,8,16,24   (2 x float4)
    const int wRow = tid >> 1;             // 0..127
    const int wCol = (tid & 1) << 4;       // 0,16        (4 x float4)

    const float* aLd = A + (long long)(m0 + aRow) * ldA + aCol;
    const float* wLd = W + (long long)(n0 + wRow) * ldW + wCol;

    v8f c00 = {0.f,0.f,0.f,0.f,0.f,0.f,0.f,0.f};
    v8f c01 = c00, c10 = c00, c11 = c00;

    const int kTiles = K >> 5;
    for (int kt = 0; kt < kTiles; ++kt) {
        const int k0 = kt << 5;
        f32x4 a4a = *(const f32x4*)(aLd + k0);
        f32x4 a4b = *(const f32x4*)(aLd + k0 + 4);
        f32x4 w40 = *(const f32x4*)(wLd + k0);
        f32x4 w41 = *(const f32x4*)(wLd + k0 + 4);
        f32x4 w42 = *(const f32x4*)(wLd + k0 + 8);
        f32x4 w43 = *(const f32x4*)(wLd + k0 + 12);
        if (kt + 1 < kTiles) {                       // gfx1250 global_prefetch
            __builtin_prefetch(aLd + k0 + 32, 0, 3);
            __builtin_prefetch(wLd + k0 + 32, 0, 3);
        }
        __syncthreads();   // previous iteration's fragment reads done
        __bf16* ap = &aT[aRow * LDSS + aCol];
        ap[0] = (__bf16)a4a.x; ap[1] = (__bf16)a4a.y;
        ap[2] = (__bf16)a4a.z; ap[3] = (__bf16)a4a.w;
        ap[4] = (__bf16)a4b.x; ap[5] = (__bf16)a4b.y;
        ap[6] = (__bf16)a4b.z; ap[7] = (__bf16)a4b.w;
        __bf16* wp = &wT[wRow * LDSS + wCol];
        wp[0]  = (__bf16)w40.x; wp[1]  = (__bf16)w40.y;
        wp[2]  = (__bf16)w40.z; wp[3]  = (__bf16)w40.w;
        wp[4]  = (__bf16)w41.x; wp[5]  = (__bf16)w41.y;
        wp[6]  = (__bf16)w41.z; wp[7]  = (__bf16)w41.w;
        wp[8]  = (__bf16)w42.x; wp[9]  = (__bf16)w42.y;
        wp[10] = (__bf16)w42.z; wp[11] = (__bf16)w42.w;
        wp[12] = (__bf16)w43.x; wp[13] = (__bf16)w43.y;
        wp[14] = (__bf16)w43.z; wp[15] = (__bf16)w43.w;
        __syncthreads();
        // A fragments (rows wm*32 + {0,16} + l), ISA bf16 A-layout K packing
        Frag a0, a1, b0, b1;
        {
            const int r0 = (wm * 32 + l)      * LDSS;
            const int r1 = (wm * 32 + 16 + l) * LDSS;
            a0.q[0] = *(const u32x4*)&aT[r0 + hi * 8];
            a0.q[1] = *(const u32x4*)&aT[r0 + 16 + hi * 8];
            a1.q[0] = *(const u32x4*)&aT[r1 + hi * 8];
            a1.q[1] = *(const u32x4*)&aT[r1 + 16 + hi * 8];
        }
        // B fragments (lane = column n of B = row n of W; cols wn*32 + {0,16})
        {
            const int r0 = (wn * 32 + l)      * LDSS;
            const int r1 = (wn * 32 + 16 + l) * LDSS;
            b0.q[0] = *(const u32x4*)&wT[r0 + hi * 8];
            b0.q[1] = *(const u32x4*)&wT[r0 + 16 + hi * 8];
            b1.q[0] = *(const u32x4*)&wT[r1 + hi * 8];
            b1.q[1] = *(const u32x4*)&wT[r1 + 16 + hi * 8];
        }
        c00 = __builtin_amdgcn_wmma_f32_16x16x32_bf16(false, a0.v, false, b0.v, (short)0, c00, false, false);
        c01 = __builtin_amdgcn_wmma_f32_16x16x32_bf16(false, a0.v, false, b1.v, (short)0, c01, false, false);
        c10 = __builtin_amdgcn_wmma_f32_16x16x32_bf16(false, a1.v, false, b0.v, (short)0, c10, false, false);
        c11 = __builtin_amdgcn_wmma_f32_16x16x32_bf16(false, a1.v, false, b1.v, (short)0, c11, false, false);
    }

    // Epilogue: C VGPR r -> (m = mbase + hi*8 + r, n = nbase + (lane&15))
    const int nA = n0 + wn * 32 + l;
    const int nB = nA + 16;
    const float bnA = bias ? bias[nA] : 0.f;
    const float bnB = bias ? bias[nB] : 0.f;
#pragma unroll
    for (int r = 0; r < 8; ++r) {
        const int mA = m0 + wm * 32 + hi * 8 + r;
        const int mB = mA + 16;
        float v00 = c00[r] * scale + bnA;
        float v01 = c01[r] * scale + bnB;
        float v10 = c10[r] * scale + bnA;
        float v11 = c11[r] * scale + bnB;
        if (act == 1) {            // SiLU
            v00 = v00 / (1.f + __expf(-v00));
            v01 = v01 / (1.f + __expf(-v01));
            v10 = v10 / (1.f + __expf(-v10));
            v11 = v11 / (1.f + __expf(-v11));
        } else if (act == 2) {     // exact GELU
            v00 = 0.5f * v00 * (1.f + erff(v00 * 0.70710678118f));
            v01 = 0.5f * v01 * (1.f + erff(v01 * 0.70710678118f));
            v10 = 0.5f * v10 * (1.f + erff(v10 * 0.70710678118f));
            v11 = 0.5f * v11 * (1.f + erff(v11 * 0.70710678118f));
        }
        const long long oA = (long long)mA * ldO;
        const long long oB = (long long)mB * ldO;
        if (Res) {
            v00 += resScale * Res[oA + nA];
            v01 += resScale * Res[oA + nB];
            v10 += resScale * Res[oB + nA];
            v11 += resScale * Res[oB + nB];
        }
        Out[oA + nA] = v00;
        Out[oA + nB] = v01;
        Out[oB + nA] = v10;
        Out[oB + nB] = v11;
    }
}

// ---------------------------------------------------------------------------
// Row LayerNorm over D=1024, one block per row.
// ---------------------------------------------------------------------------
__global__ __launch_bounds__(256)
void layernorm_kernel(const float* __restrict__ X, const float* __restrict__ gw,
                      const float* __restrict__ bw, float* __restrict__ Y)
{
    __shared__ float sred[8];
    __shared__ float sbc;
    const long long row = blockIdx.x;
    const float* x = X + row * DMODEL;
    float* y = Y + row * DMODEL;
    const int tid = threadIdx.x;

    float v0 = x[tid], v1 = x[tid + 256], v2 = x[tid + 512], v3 = x[tid + 768];
    float s = v0 + v1 + v2 + v3;
    for (int o = 16; o > 0; o >>= 1) s += __shfl_down(s, o, 32);
    if ((tid & 31) == 0) sred[tid >> 5] = s;
    __syncthreads();
    if (tid == 0) { float t = 0.f; for (int i = 0; i < 8; ++i) t += sred[i]; sbc = t * (1.f / DMODEL); }
    __syncthreads();
    const float m = sbc;
    const float d0 = v0 - m, d1 = v1 - m, d2 = v2 - m, d3 = v3 - m;
    float q = d0 * d0 + d1 * d1 + d2 * d2 + d3 * d3;
    for (int o = 16; o > 0; o >>= 1) q += __shfl_down(q, o, 32);
    if ((tid & 31) == 0) sred[tid >> 5] = q;
    __syncthreads();
    if (tid == 0) { float t = 0.f; for (int i = 0; i < 8; ++i) t += sred[i]; sbc = t * (1.f / DMODEL); }
    __syncthreads();
    const float inv = rsqrtf(sbc + 1e-5f);
    y[tid]       = d0 * inv * gw[tid]       + bw[tid];
    y[tid + 256] = d1 * inv * gw[tid + 256] + bw[tid + 256];
    y[tid + 512] = d2 * inv * gw[tid + 512] + bw[tid + 512];
    y[tid + 768] = d3 * inv * gw[tid + 768] + bw[tid + 768];
}

// ---------------------------------------------------------------------------
// Row softmax over 1024 columns, one block per row (in place).
// ---------------------------------------------------------------------------
__global__ __launch_bounds__(256)
void softmax_kernel(float* __restrict__ P)
{
    __shared__ float sred[8];
    __shared__ float sbc;
    const long long row = blockIdx.x;
    float* p = P + row * 1024;
    const int tid = threadIdx.x;

    float v0 = p[tid], v1 = p[tid + 256], v2 = p[tid + 512], v3 = p[tid + 768];
    float mx = fmaxf(fmaxf(v0, v1), fmaxf(v2, v3));
    for (int o = 16; o > 0; o >>= 1) mx = fmaxf(mx, __shfl_down(mx, o, 32));
    if ((tid & 31) == 0) sred[tid >> 5] = mx;
    __syncthreads();
    if (tid == 0) { float t = sred[0]; for (int i = 1; i < 8; ++i) t = fmaxf(t, sred[i]); sbc = t; }
    __syncthreads();
    const float M = sbc;
    v0 = __expf(v0 - M); v1 = __expf(v1 - M); v2 = __expf(v2 - M); v3 = __expf(v3 - M);
    float s = v0 + v1 + v2 + v3;
    for (int o = 16; o > 0; o >>= 1) s += __shfl_down(s, o, 32);
    if ((tid & 31) == 0) sred[tid >> 5] = s;
    __syncthreads();
    if (tid == 0) { float t = 0.f; for (int i = 0; i < 8; ++i) t += sred[i]; sbc = t; }
    __syncthreads();
    const float inv = 1.f / sbc;
    p[tid] = v0 * inv; p[tid + 256] = v1 * inv; p[tid + 512] = v2 * inv; p[tid + 768] = v3 * inv;
}

// ---------------------------------------------------------------------------
// Embedding + positional:  X[b,s,:] = emb[trg[b,s],:] * sqrt(D) + pos[s,:]
// ---------------------------------------------------------------------------
__global__ __launch_bounds__(256)
void embed_kernel(const int* __restrict__ trg, const float* __restrict__ emb,
                  const float* __restrict__ pos, float* __restrict__ X)
{
    const long long idx = (long long)blockIdx.x * 256 + threadIdx.x;
    if (idx >= (long long)NTOK * DMODEL) return;
    const int d = (int)(idx & (DMODEL - 1));
    const long long t = idx >> 10;
    const int s = (int)(t & (SEQ - 1));
    const int tok = trg[t];
    X[idx] = emb[(long long)tok * DMODEL + d] * 32.0f + pos[(long long)s * DMODEL + d];
}

// ---------------------------------------------------------------------------
// Mamba selective scan. One thread per (batch, channel); 16-wide state in regs.
// pr holds silu(xn@Wp+bp); delta = sigmoid(pr[:, :DI]), gate = pr[:, DI:].
// ---------------------------------------------------------------------------
__global__ __launch_bounds__(256)
void mamba_scan_kernel(const float* __restrict__ xup, const float* __restrict__ pr,
                       const float* __restrict__ Am,  const float* __restrict__ Bm,
                       const float* __restrict__ Cm,  const float* __restrict__ Dm,
                       float* __restrict__ ys)
{
    const int c = blockIdx.x * 256 + threadIdx.x;   // 0..DI-1
    const int b = blockIdx.y;
    float a[NDST], bb[NDST], cc[NDST], h[NDST];
#pragma unroll
    for (int i = 0; i < NDST; ++i) {
        a[i]  = Am[c * NDST + i];
        bb[i] = Bm[c * NDST + i];
        cc[i] = Cm[c * NDST + i];
        h[i]  = 0.f;
    }
    const float dsk = Dm[c];
    const float* xu = xup + (long long)b * SEQ * DI + c;
    const float* pd = pr  + (long long)b * SEQ * (2 * DI) + c;
    const float* pg = pd + DI;
    float* yo = ys + (long long)b * SEQ * DI + c;

    for (int s = 0; s < SEQ; ++s) {
        const float u  = xu[(long long)s * DI];
        const float dl = pd[(long long)s * 2 * DI];
        const float g  = pg[(long long)s * 2 * DI];
        const float d  = 1.f / (1.f + __expf(-dl));
        float y = 0.f;
#pragma unroll
        for (int i = 0; i < NDST; ++i) {
            h[i] = h[i] * __expf(d * a[i]) + u * bb[i];
            y += h[i] * cc[i];
        }
        y += u * dsk;
        yo[(long long)s * DI] = y * (1.f / (1.f + __expf(-g)));
    }
}

// ---------------------------------------------------------------------------
// V [B,S,H,hd] -> VT [B,H,hd,S]   (index by output, coalesced writes)
// ---------------------------------------------------------------------------
__global__ __launch_bounds__(256)
void transpose_v_kernel(const float* __restrict__ V, float* __restrict__ VT)
{
    const long long o = (long long)blockIdx.x * 256 + threadIdx.x;
    if (o >= (long long)NTOK * DMODEL) return;
    const int s = (int)(o & (SEQ - 1));
    const int d = (int)((o >> 10) & (HDIM - 1));
    const int h = (int)((o >> 17) & (NHEADS - 1));
    const int b = (int)(o >> 20);
    VT[o] = V[((long long)(b * SEQ + s)) * DMODEL + h * HDIM + d];
}

// ---------------------------------------------------------------------------
// OB [B,H,S,hd] -> OM [B,S,H*hd]
// ---------------------------------------------------------------------------
__global__ __launch_bounds__(256)
void merge_o_kernel(const float* __restrict__ OB, float* __restrict__ OM)
{
    const long long o = (long long)blockIdx.x * 256 + threadIdx.x;
    if (o >= (long long)NTOK * DMODEL) return;
    const int d = (int)(o & (HDIM - 1));
    const int h = (int)((o >> 7) & (NHEADS - 1));
    const int s = (int)((o >> 10) & (SEQ - 1));
    const int b = (int)(o >> 20);
    OM[o] = OB[((long long)((b * NHEADS + h) * SEQ + s)) * HDIM + d];
}

// ---------------------------------------------------------------------------
// Host-side launch helpers
// ---------------------------------------------------------------------------
static void gemm(hipStream_t st,
                 const float* A, int ldA, long long aSb, long long aSh,
                 const float* W, int ldW, long long wSb, long long wSh,
                 const float* bias,
                 float* Out, int ldO, long long oSb, long long oSh,
                 const float* Res, float resScale,
                 int M, int N, int K, float scale, int act, int zDiv, int Z)
{
    dim3 grid(N / 128, M / 64, Z);
    hipLaunchKernelGGL(wmma_gemm_kernel, grid, dim3(256), 0, st,
                       A, ldA, aSb, aSh, W, ldW, wSb, wSh, bias,
                       Out, ldO, oSb, oSh, Res, resScale, M, N, K, scale, act, zDiv);
}

// Per-layer leaf order (jax tree_leaves: dict keys sorted at each level)
enum {
    L_W1 = 0, L_W2, L_WIN, L_WOUT, L_BIN, L_BOUT, L_B1, L_B2,
    L_LN1B, L_LN1G, L_LN2B, L_LN2G,
    L_MA, L_MB, L_MC, L_MD, L_MWDOWN, L_MWP, L_MWUP, L_MBDOWN, L_MBP, L_MBUP,
    L_MLNB, L_MLNG, LEAVES_PER_LAYER
};

extern "C" void kernel_launch(void* const* d_in, const int* in_sizes, int n_in,
                              void* d_out, int out_size, void* d_ws, size_t ws_size,
                              hipStream_t stream)
{
    if (n_in < 3 + 4 * LEAVES_PER_LAYER + 3) return;

    const int*   trg    = (const int*)d_in[0];
    const float* memory = (const float*)d_in[1];
    const float* emb    = (const float*)d_in[2];
    const float* lnf_b  = (const float*)d_in[3 + 4 * LEAVES_PER_LAYER + 0];
    const float* lnf_g  = (const float*)d_in[3 + 4 * LEAVES_PER_LAYER + 1];
    const float* pos    = (const float*)d_in[3 + 4 * LEAVES_PER_LAYER + 2];
    auto LP = [&](int l, int j) -> const float* {
        return (const float*)d_in[3 + l * LEAVES_PER_LAYER + j];
    };

    // Workspace layout (floats); T4 = 4096*1024
    float* ws = (float*)d_ws;
    const size_t T4 = (size_t)NTOK * DMODEL;
    float* XN  = ws;                 // 1*T4  layernorm output
    float* XUP = XN  + T4;           // 2*T4  mamba up-projection
    float* PR  = XUP + 2 * T4;       // 4*T4  mamba gate/delta (silu'd); aliased as FFN hidden
    float* YS  = PR  + 4 * T4;       // 2*T4  scan output
    float* Qb  = YS  + 2 * T4;       // 1*T4
    float* Kb  = Qb  + T4;           // 1*T4
    float* Vb  = Kb  + T4;           // 1*T4
    float* Pm  = Vb  + T4;           // 8*T4  attention scores [B*H,1024,1024]
    float* VT  = Pm  + 8 * T4;       // 1*T4  V transposed per head
    float* OB  = VT  + T4;           // 1*T4  attention out [B,H,S,hd]
    float* OM  = OB  + T4;           // 1*T4  merged heads
    float* X   = OM  + T4;           // 1*T4  residual stream
    (void)ws_size; (void)in_sizes; (void)out_size;

    const long long SD = (long long)SEQ * DMODEL;       // 1048576
    const dim3 b256(256);
    const int nElemBlocks = (int)((T4 + 255) / 256);

    hipLaunchKernelGGL(embed_kernel, dim3(nElemBlocks), b256, 0, stream, trg, emb, pos, X);

    for (int l = 0; l < 4; ++l) {
        // ---- Mamba block (internal residual + outer decoder residual => 2*X) ----
        hipLaunchKernelGGL(layernorm_kernel, dim3(NTOK), b256, 0, stream,
                           X, LP(l, L_MLNG), LP(l, L_MLNB), XN);
        gemm(stream, XN, DMODEL, 0, 0, LP(l, L_MWUP), DMODEL, 0, 0, LP(l, L_MBUP),
             XUP, DI, 0, 0, nullptr, 0.f, NTOK, DI, DMODEL, 1.f, 0, 1, 1);
        gemm(stream, XN, DMODEL, 0, 0, LP(l, L_MWP), DMODEL, 0, 0, LP(l, L_MBP),
             PR, 2 * DI, 0, 0, nullptr, 0.f, NTOK, 2 * DI, DMODEL, 1.f, 1, 1, 1);
        hipLaunchKernelGGL(mamba_scan_kernel, dim3(DI / 256, BATCH), b256, 0, stream,
                           XUP, PR, LP(l, L_MA), LP(l, L_MB), LP(l, L_MC), LP(l, L_MD), YS);
        gemm(stream, YS, DI, 0, 0, LP(l, L_MWDOWN), DI, 0, 0, LP(l, L_MBDOWN),
             X, DMODEL, 0, 0, X, 2.0f, NTOK, DMODEL, DI, 1.f, 0, 1, 1);

        // ---- Cross attention ----
        hipLaunchKernelGGL(layernorm_kernel, dim3(NTOK), b256, 0, stream,
                           X, LP(l, L_LN1G), LP(l, L_LN1B), XN);
        const float* Win = LP(l, L_WIN);
        const float* bin = LP(l, L_BIN);
        gemm(stream, XN, DMODEL, 0, 0, Win, DMODEL, 0, 0, bin,
             Qb, DMODEL, 0, 0, nullptr, 0.f, NTOK, DMODEL, DMODEL, 1.f, 0, 1, 1);
        gemm(stream, memory, DMODEL, 0, 0, Win + (long long)DMODEL * DMODEL, DMODEL, 0, 0,
             bin + DMODEL, Kb, DMODEL, 0, 0, nullptr, 0.f, NTOK, DMODEL, DMODEL, 1.f, 0, 1, 1);
        gemm(stream, memory, DMODEL, 0, 0, Win + 2LL * DMODEL * DMODEL, DMODEL, 0, 0,
             bin + 2 * DMODEL, Vb, DMODEL, 0, 0, nullptr, 0.f, NTOK, DMODEL, DMODEL, 1.f, 0, 1, 1);
        // scores: P[z] = Q_z @ K_z^T / sqrt(hd), z = b*8+h
        gemm(stream, Qb, DMODEL, SD, HDIM, Kb, DMODEL, SD, HDIM, nullptr,
             Pm, SEQ, 8LL * SD, SD, nullptr, 0.f,
             SEQ, SEQ, HDIM, 0.08838834764831845f, 0, NHEADS, BATCH * NHEADS);
        hipLaunchKernelGGL(softmax_kernel, dim3(BATCH * NHEADS * SEQ), b256, 0, stream, Pm);
        hipLaunchKernelGGL(transpose_v_kernel, dim3(nElemBlocks), b256, 0, stream, Vb, VT);
        // O[z] = P[z] @ V[z]  (W = V^T per head, [hd, S])
        gemm(stream, Pm, SEQ, 8LL * SD, SD, VT, SEQ, 8LL * HDIM * SEQ, (long long)HDIM * SEQ,
             nullptr, OB, HDIM, 8LL * HDIM * SEQ, (long long)HDIM * SEQ, nullptr, 0.f,
             SEQ, HDIM, SEQ, 1.f, 0, NHEADS, BATCH * NHEADS);
        hipLaunchKernelGGL(merge_o_kernel, dim3(nElemBlocks), b256, 0, stream, OB, OM);
        gemm(stream, OM, DMODEL, 0, 0, LP(l, L_WOUT), DMODEL, 0, 0, LP(l, L_BOUT),
             X, DMODEL, 0, 0, X, 1.0f, NTOK, DMODEL, DMODEL, 1.f, 0, 1, 1);

        // ---- FFN ----
        hipLaunchKernelGGL(layernorm_kernel, dim3(NTOK), b256, 0, stream,
                           X, LP(l, L_LN2G), LP(l, L_LN2B), XN);
        gemm(stream, XN, DMODEL, 0, 0, LP(l, L_W1), DMODEL, 0, 0, LP(l, L_B1),
             PR, 4 * DMODEL, 0, 0, nullptr, 0.f, NTOK, 4 * DMODEL, DMODEL, 1.f, 2, 1, 1);
        gemm(stream, PR, 4 * DMODEL, 0, 0, LP(l, L_W2), 4 * DMODEL, 0, 0, LP(l, L_B2),
             X, DMODEL, 0, 0, X, 1.0f, NTOK, DMODEL, 4 * DMODEL, 1.f, 0, 1, 1);
    }

    hipLaunchKernelGGL(layernorm_kernel, dim3(NTOK), b256, 0, stream,
                       X, lnf_g, lnf_b, (float*)d_out);
}
